// GRUSeqDecoder_53644141527218
// MI455X (gfx1250) — compile-verified, matrix operand order checked by reading, not run on hardware
//
#include <hip/hip_runtime.h>
#include <hip/hip_bf16.h>
#include <stdint.h>

typedef __attribute__((ext_vector_type(16))) _Float16 v16h;
typedef __attribute__((ext_vector_type(8)))  _Float16 v8h;
typedef __attribute__((ext_vector_type(8)))  float    v8f;

#define Bz 8
#define Tz 64
#define Lz 128
#define Hz 512
#define Ez 512
#define Vz 32000
#define H3 1536

#if defined(__has_builtin)
#if __has_builtin(__builtin_amdgcn_tanhf)
#define TANHF __builtin_amdgcn_tanhf
#else
#define TANHF tanhf
#endif
#else
#define TANHF tanhf
#endif

// ---------------------------------------------------------------------------
// WMMA helpers (CDNA5 wave32, v_wmma_f32_16x16x32_f16)
// ---------------------------------------------------------------------------
__device__ __forceinline__ v8f wmma16(v16h a, v16h b, v8f c) {
  return __builtin_amdgcn_wmma_f32_16x16x32_f16(false, a, false, b, (short)0, c,
                                                false, false);
}

// Fragment (16x32 f16) from a per-lane pre-offset pointer:
// p = base + (row0 + lane%16)*ld + (lane/16)*8 + k0. Two 16B loads.
__device__ __forceinline__ v16h frag_from(const _Float16* p) {
  v8h lo = *(const v8h*)(p);
  v8h hi = *(const v8h*)(p + 16);
  v16h r;
#pragma unroll
  for (int i = 0; i < 8; ++i) { r[i] = lo[i]; r[i + 8] = hi[i]; }
  return r;
}

__device__ __forceinline__ float wave_sum(float x) {
#pragma unroll
  for (int o = 16; o > 0; o >>= 1) x += __shfl_xor(x, o, 32);
  return x;
}
__device__ __forceinline__ float wave_max(float x) {
#pragma unroll
  for (int o = 16; o > 0; o >>= 1) x = fmaxf(x, __shfl_xor(x, o, 32));
  return x;
}

// ---------------------------------------------------------------------------
// f32 -> f16 convert (grid-stride)
// ---------------------------------------------------------------------------
__global__ void cvt_kernel(const float* __restrict__ s, _Float16* __restrict__ d,
                           long n) {
  long stride = (long)gridDim.x * blockDim.x;
  for (long i = (long)blockIdx.x * blockDim.x + threadIdx.x; i < n; i += stride)
    d[i] = (_Float16)s[i];
}

// mem [B,L,H] f32  ->  memT [B,H,L] f16   (for the ctx GEMM B operand)
__global__ void transpose_mem_kernel(const float* __restrict__ mem,
                                     _Float16* __restrict__ memT) {
  int bl = blockIdx.x;  // b*Lz + l
  int b = bl >> 7, l = bl & 127;
  const float* src = mem + (long)bl * Hz;
  for (int h = threadIdx.x; h < Hz; h += blockDim.x)
    memT[((long)b * Hz + h) * Lz + l] = (_Float16)src[h];
}

// ---------------------------------------------------------------------------
// Embedding lookup (teacher forcing: t=0 -> BOS, else gt[t-1])
// ---------------------------------------------------------------------------
__global__ void embed_kernel(const int* __restrict__ gt,
                             const int* __restrict__ BOSp,
                             const float* __restrict__ emb,
                             float* __restrict__ x32,
                             _Float16* __restrict__ x16) {
  int row = blockIdx.x;  // b*Tz + t
  int b = row >> 6, t = row & 63;
  int idx = (t == 0) ? BOSp[0] : gt[b * Tz + t - 1];
  const float* e = emb + (long)idx * Ez;
  for (int c = threadIdx.x; c < Ez; c += blockDim.x) {
    float v = e[c];
    x32[(long)row * Ez + c] = v;
    x16[(long)row * Ez + c] = (_Float16)v;
  }
}

// ---------------------------------------------------------------------------
// Batched WMMA GEMM: C[M,N] = A[M,K] @ W[N,K]^T (+bias), f16 in / f32 out.
// block = 256 (8 waves); each wave owns 4 M-tiles x 1 N-tile so each B
// fragment is reused 4x (4 back-to-back WMMAs per k-step). Four independent
// A-row pointers act as induction variables so all 10 loads per k-step are
// address-ready and can be clustered.
// Requires M % 64 == 0, N % 16 == 0, K % 32 == 0, lda/ldb % 8 == 0.
// ---------------------------------------------------------------------------
__global__ void gemm_f16_kernel(const _Float16* __restrict__ A,
                                const _Float16* __restrict__ Bm,
                                float* __restrict__ C,
                                _Float16* __restrict__ C16,
                                const float* __restrict__ bias,
                                int M, int N, int K, int lda, int ldb, int ldc,
                                long sA, long sB, long sC) {
  int wave = threadIdx.x >> 5, lane = threadIdx.x & 31;
  int ntile = blockIdx.x * 8 + wave;
  if (ntile * 16 >= N) return;  // wave-uniform exit (EXEC stays all-ones)
  int m0 = blockIdx.y * 64;
  int z = blockIdx.z;
  A += (long)z * sA; Bm += (long)z * sB; C += (long)z * sC;
  if (C16) C16 += (long)z * sC;

  int mi = lane & 15, kh = lane >> 4;
  const _Float16* pB = Bm + (long)(ntile * 16 + mi) * ldb + kh * 8;
  const _Float16* pA0 = A + (long)(m0 + mi) * lda + kh * 8;
  const _Float16* pA1 = pA0 + 16 * lda;
  const _Float16* pA2 = pA0 + 32 * lda;
  const _Float16* pA3 = pA0 + 48 * lda;

  v8f acc0 = {}, acc1 = {}, acc2 = {}, acc3 = {};
  for (int k0 = 0; k0 < K; k0 += 32) {
    v16h bf = frag_from(pB);
    v16h a0 = frag_from(pA0);
    v16h a1 = frag_from(pA1);
    v16h a2 = frag_from(pA2);
    v16h a3 = frag_from(pA3);
    pB += 32; pA0 += 32; pA1 += 32; pA2 += 32; pA3 += 32;
    acc0 = wmma16(a0, bf, acc0);
    acc1 = wmma16(a1, bf, acc1);
    acc2 = wmma16(a2, bf, acc2);
    acc3 = wmma16(a3, bf, acc3);
  }

  int n = ntile * 16 + mi;
  float bv = bias ? bias[n] : 0.0f;
  v8f accs[4] = {acc0, acc1, acc2, acc3};
  if (C16) {
#pragma unroll
    for (int tt = 0; tt < 4; ++tt) {
      int mbase = m0 + tt * 16 + kh * 8;
#pragma unroll
      for (int r = 0; r < 8; ++r) {
        float v = accs[tt][r] + bv;
        long idx = (long)(mbase + r) * ldc + n;
        C[idx] = v;
        C16[idx] = (_Float16)v;
      }
    }
  } else {
#pragma unroll
    for (int tt = 0; tt < 4; ++tt) {
      int mbase = m0 + tt * 16 + kh * 8;
#pragma unroll
      for (int r = 0; r < 8; ++r)
        C[(long)(mbase + r) * ldc + n] = accs[tt][r] + bv;
    }
  }
}

// ---------------------------------------------------------------------------
// GRU scan over T for one layer. Single workgroup, 1024 threads (32 waves).
// h (f16, padded to 16 rows) and hg (f32) live in LDS; hg = h @ Whh^T + bhh
// via WMMA (96 N-tiles = 32 waves x 3).
// ---------------------------------------------------------------------------
__global__ void __launch_bounds__(1024)
gru_scan_kernel(const _Float16* __restrict__ Wh,   // [3H, H] f16
                const float* __restrict__ bhh,     // [3H]
                const float* __restrict__ xg,      // [B*T, 3H]
                const float* __restrict__ h0,      // + b*h0_stride
                int h0_stride,
                _Float16* __restrict__ y16,        // [B*T, H]
                float* __restrict__ y32) {         // nullable [B*T, H]
  __shared__ __align__(16) _Float16 hL[16 * Hz];   // 16 KB
  __shared__ float hgL[8 * H3];                    // 48 KB
  int tid = threadIdx.x, wave = tid >> 5, lane = tid & 31;
  int mi = lane & 15, kh = lane >> 4;
  int e0 = tid * 4, b = e0 >> 9, i0 = e0 & 511;

  float hreg[4];
#pragma unroll
  for (int j = 0; j < 4; ++j) {
    float hv = h0[b * h0_stride + i0 + j];
    hreg[j] = hv;
    hL[b * Hz + i0 + j] = (_Float16)hv;
  }
  for (int j = tid; j < 8 * Hz; j += 1024) hL[8 * Hz + j] = (_Float16)0.0f;
  __syncthreads();

  const _Float16* pH = hL + mi * Hz + kh * 8;  // per-lane A base (LDS)

  for (int t = 0; t < Tz; ++t) {
#pragma unroll
    for (int q = 0; q < 3; ++q) {
      int nt = wave * 3 + q;
      const _Float16* pW = Wh + (long)(nt * 16 + mi) * Hz + kh * 8;
      v8f acc = {};
      for (int k0 = 0; k0 < Hz; k0 += 32) {
        v16h a = frag_from(pH + k0);
        v16h bf = frag_from(pW + k0);
        acc = wmma16(a, bf, acc);
      }
      if (lane < 16) {  // rows 0..7 are the real batch rows
        int n = nt * 16 + lane;
        float bb = bhh[n];
#pragma unroll
        for (int r = 0; r < 8; ++r) hgL[r * H3 + n] = acc[r] + bb;
      }
    }
    __syncthreads();

    const float* xr = xg + (long)(b * Tz + t) * H3;
#pragma unroll
    for (int j = 0; j < 4; ++j) {
      int i = i0 + j;
      float hr = hgL[b * H3 + i];
      float hz = hgL[b * H3 + Hz + i];
      float hn = hgL[b * H3 + 2 * Hz + i];
      float r = 1.0f / (1.0f + __expf(-(xr[i] + hr)));
      float z = 1.0f / (1.0f + __expf(-(xr[Hz + i] + hz)));
      float nn = TANHF(xr[2 * Hz + i] + r * hn);
      float h = (1.0f - z) * nn + z * hreg[j];
      hreg[j] = h;
      long o = (long)(b * Tz + t) * Hz + i;
      y16[o] = (_Float16)h;
      if (y32) y32[o] = h;
      hL[b * Hz + i] = (_Float16)h;
    }
    __syncthreads();
  }
}

// ---------------------------------------------------------------------------
// Attention scores: s[b,t,l] = sum_h v[h] * tanh(qh[b,t,h] + kh[b,l,h]).
// One wave per (b,t,l); 8 waves / block. Uses hardware v_tanh_f32.
// ---------------------------------------------------------------------------
__global__ void attn_score_kernel(const float* __restrict__ qh,
                                  const float* __restrict__ kh,
                                  const float* __restrict__ v,
                                  float* __restrict__ s) {
  int gw = (blockIdx.x * blockDim.x + threadIdx.x) >> 5;
  int lane = threadIdx.x & 31;
  int b = gw >> 13;  // / (Tz*Lz)
  int r = gw & 8191;
  int t = r >> 7, l = r & 127;
  const float* q = qh + (long)(b * Tz + t) * Hz + lane;
  const float* k = kh + (long)(b * Lz + l) * Hz + lane;
  const float* vv = v + lane;
  float acc = 0.0f;
#pragma unroll
  for (int j = 0; j < 16; ++j)
    acc += vv[j * 32] * TANHF(q[j * 32] + k[j * 32]);
  acc = wave_sum(acc);
  if (lane == 0) s[gw] = acc;
}

// ---------------------------------------------------------------------------
// Masked softmax over L + copy-probability normalization. Block = 128 (= L).
// ---------------------------------------------------------------------------
__global__ void attn_softmax_kernel(const float* __restrict__ s,
                                    const unsigned char* __restrict__ mask,
                                    const float* __restrict__ w,
                                    float* __restrict__ p,
                                    _Float16* __restrict__ a16) {
  __shared__ float red[4];
  int row = blockIdx.x;  // b*Tz + t
  int b = row >> 6;
  int l = threadIdx.x, wave = l >> 5, lane = l & 31;
  bool ok = mask[b * Lz + l] != 0;
  float sv = ok ? s[(long)row * Lz + l] : -INFINITY;

  float m = wave_max(sv);
  if (lane == 0) red[wave] = m;
  __syncthreads();
  float mx = fmaxf(fmaxf(red[0], red[1]), fmaxf(red[2], red[3]));
  __syncthreads();

  float ev = ok ? __expf(sv - mx) : 0.0f;
  float sm = wave_sum(ev);
  if (lane == 0) red[wave] = sm;
  __syncthreads();
  float tot = red[0] + red[1] + red[2] + red[3];
  __syncthreads();

  float a = ev / tot;
  float pv = w[b * Lz + l] * a;
  float ps = wave_sum(pv);
  if (lane == 0) red[wave] = ps;
  __syncthreads();
  float ptot = red[0] + red[1] + red[2] + red[3];
  pv = pv / (1e-8f + ptot);

  a16[(long)row * Lz + l] = (_Float16)a;
  p[(long)row * Lz + l] = pv;
}

// ---------------------------------------------------------------------------
// feat16[b,t,:] = concat(c0, c1, x, dec) (2048 wide, f16)
// ---------------------------------------------------------------------------
__global__ void feat_kernel(const float* __restrict__ c0,
                            const float* __restrict__ c1,
                            const _Float16* __restrict__ x16,
                            const _Float16* __restrict__ dec16,
                            _Float16* __restrict__ feat) {
  int row = blockIdx.x;
  for (int c = threadIdx.x; c < 2048; c += blockDim.x) {
    _Float16 v;
    if (c < 512)        v = (_Float16)c0[(long)row * Hz + c];
    else if (c < 1024)  v = (_Float16)c1[(long)row * Hz + (c - 512)];
    else if (c < 1536)  v = x16[(long)row * Ez + (c - 1024)];
    else                v = dec16[(long)row * Hz + (c - 1536)];
    feat[(long)row * 2048 + c] = v;
  }
}

// ---------------------------------------------------------------------------
// pm = softmax(feat @ mix_W^T + mix_b) over 3. One wave per row.
// ---------------------------------------------------------------------------
__global__ void mix_kernel(const _Float16* __restrict__ feat,
                           const float* __restrict__ mixW,
                           const float* __restrict__ mixb,
                           float* __restrict__ pm) {
  int wave = threadIdx.x >> 5, lane = threadIdx.x & 31;
  int row = blockIdx.x * 8 + wave;
  float g[3];
#pragma unroll
  for (int gg = 0; gg < 3; ++gg) {
    float acc = 0.0f;
    for (int j = 0; j < 64; ++j) {
      int k = j * 32 + lane;
      acc += (float)feat[(long)row * 2048 + k] * mixW[gg * 2048 + k];
    }
    g[gg] = wave_sum(acc) + mixb[gg];
  }
  if (lane == 0) {
    float mx = fmaxf(g[0], fmaxf(g[1], g[2]));
    float e0 = __expf(g[0] - mx), e1 = __expf(g[1] - mx), e2 = __expf(g[2] - mx);
    float inv = 1.0f / (e0 + e1 + e2);
    pm[row * 3 + 0] = e0 * inv;
    pm[row * 3 + 1] = e1 * inv;
    pm[row * 3 + 2] = e2 * inv;
  }
}

// ---------------------------------------------------------------------------
// In-place row softmax over V=32000. Block = 256 (8 waves).
// ---------------------------------------------------------------------------
__global__ void row_softmax_kernel(float* __restrict__ X) {
  __shared__ float red[8];
  int row = blockIdx.x;
  float* x = X + (long)row * Vz;
  int tid = threadIdx.x, wave = tid >> 5, lane = tid & 31;

  float m = -INFINITY;
  for (int i = tid; i < Vz; i += 256) m = fmaxf(m, x[i]);
  m = wave_max(m);
  if (lane == 0) red[wave] = m;
  __syncthreads();
  float mx = red[0];
#pragma unroll
  for (int i = 1; i < 8; ++i) mx = fmaxf(mx, red[i]);
  __syncthreads();

  float s = 0.0f;
  for (int i = tid; i < Vz; i += 256) s += __expf(x[i] - mx);
  s = wave_sum(s);
  if (lane == 0) red[wave] = s;
  __syncthreads();
  float tot = 0.0f;
#pragma unroll
  for (int i = 0; i < 8; ++i) tot += red[i];
  float inv = 1.0f / tot;

  for (int i = tid; i < Vz; i += 256) x[i] = __expf(x[i] - mx) * inv;
}

// ---------------------------------------------------------------------------
// wcat16[b,t,:] = [pm1*p0, pm2*p1]  (2L = 256 wide)
// ---------------------------------------------------------------------------
__global__ void wcat_kernel(const float* __restrict__ p0,
                            const float* __restrict__ p1,
                            const float* __restrict__ pm,
                            _Float16* __restrict__ wcat) {
  int row = blockIdx.x, lp = threadIdx.x;
  float v = (lp < Lz) ? pm[row * 3 + 1] * p0[(long)row * Lz + lp]
                      : pm[row * 3 + 2] * p1[(long)row * Lz + (lp - Lz)];
  wcat[(long)row * 2 * Lz + lp] = (_Float16)v;
}

// ---------------------------------------------------------------------------
// ext[b,t,v] = pm0 * gen_out + wcat @ source_map.
// Per-batch 64x256x32000 GEMM; source_map (262 MB, the HBM-roofline stream)
// is converted f32->f16 on the fly with per-lane base + constant offsets and
// prefetched 2 k-steps ahead; wcat is staged to LDS with the CDNA5 async
// data mover (global_load_async_to_lds_b128 + s_wait_asynccnt).
// ---------------------------------------------------------------------------
__global__ void dist2_ext_kernel(const _Float16* __restrict__ wcat,
                                 const float* __restrict__ smap,
                                 const float* __restrict__ pm,
                                 const float* __restrict__ gen,
                                 float* __restrict__ ext) {
  __shared__ __align__(16) _Float16 aL[Tz * 2 * Lz];  // 64x256 f16 = 32 KB
  int b = blockIdx.z;
  int tid = threadIdx.x, wave = tid >> 5, lane = tid & 31;

  // Async stage of this batch's wcat tile into LDS (8 x b128 per thread).
  {
    const _Float16* gsrc = wcat + (long)b * Tz * 2 * Lz;
#pragma unroll
    for (int j = 0; j < 8; ++j) {
      int elem = (tid + j * 256) * 8;
      unsigned lds = (unsigned)(uintptr_t)(aL + elem);
      unsigned long long ga = (unsigned long long)(uintptr_t)(gsrc + elem);
      asm volatile("global_load_async_to_lds_b128 %0, %1, off"
                   :: "v"(lds), "v"(ga) : "memory");
    }
    asm volatile("s_wait_asynccnt 0x0" ::: "memory");
  }
  __syncthreads();

  int mi = lane & 15, kh = lane >> 4;
  int n0 = (blockIdx.x * 8 + wave) * 16;
  // per-lane source_map base: column n0+mi, k-rows kh*8 + {0..7,16..23}
  const float* pB = smap + (long)b * 2 * Lz * Vz + (long)kh * 8 * Vz + n0 + mi;
  const _Float16* pAL = aL + mi * (2 * Lz) + kh * 8;

  v8f acc[4] = {{}, {}, {}, {}};
  for (int k0 = 0; k0 < 2 * Lz; k0 += 32) {
    v16h bf;
#pragma unroll
    for (int i = 0; i < 16; ++i) {
      int rem = (i < 8) ? i : (i + 8);  // constant * Vz -> imm offsets
      bf[i] = (_Float16)pB[(long)rem * Vz];
    }
    __builtin_prefetch(pB + (long)64 * Vz, 0, 1);  // 2 k-steps ahead
    pB += (long)32 * Vz;
#pragma unroll
    for (int mt = 0; mt < 4; ++mt) {
      v16h a = frag_from(pAL + mt * 16 * (2 * Lz) + k0);
      acc[mt] = wmma16(a, bf, acc[mt]);
    }
  }
#pragma unroll
  for (int mt = 0; mt < 4; ++mt) {
    int tbase = mt * 16 + kh * 8;
#pragma unroll
    for (int r = 0; r < 8; ++r) {
      int t = tbase + r;
      int row = b * Tz + t;
      long idx = (long)row * Vz + n0 + mi;
      ext[idx] = acc[mt][r] + pm[row * 3 + 0] * gen[idx];
    }
  }
}

__global__ void gt_copy_kernel(const int* __restrict__ gt,
                               float* __restrict__ out) {
  int i = blockIdx.x * blockDim.x + threadIdx.x;
  if (i < Bz * Tz) out[i] = (float)gt[i];
}

// ---------------------------------------------------------------------------
// Host orchestration
// ---------------------------------------------------------------------------
extern "C" void kernel_launch(void* const* d_in, const int* in_sizes, int n_in,
                              void* d_out, int out_size, void* d_ws,
                              size_t ws_size, hipStream_t stream) {
  const float* mem0 = (const float*)d_in[0];
  const float* mem1 = (const float*)d_in[1];
  const float* w0 = (const float*)d_in[2];
  const float* w1 = (const float*)d_in[3];
  const unsigned char* mask0 = (const unsigned char*)d_in[4];
  const unsigned char* mask1 = (const unsigned char*)d_in[5];
  const float* smap = (const float*)d_in[6];
  const int* gt = (const int*)d_in[7];
  const float* init_state = (const float*)d_in[8];
  const float* emb = (const float*)d_in[9];
  const float* Wih0 = (const float*)d_in[10];
  const float* Whh0 = (const float*)d_in[11];
  const float* bih0 = (const float*)d_in[12];
  const float* bhh0 = (const float*)d_in[13];
  const float* Wih1 = (const float*)d_in[14];
  const float* Whh1 = (const float*)d_in[15];
  const float* bih1 = (const float*)d_in[16];
  const float* bhh1 = (const float*)d_in[17];
  const float* attn_Wk = (const float*)d_in[18];
  const float* attn_Wq = (const float*)d_in[19];
  const float* attn_bq = (const float*)d_in[20];
  const float* attn_v = (const float*)d_in[21];
  const float* gen_W1 = (const float*)d_in[22];
  const float* gen_b1 = (const float*)d_in[23];
  const float* gen_W2 = (const float*)d_in[24];
  const float* mix_W = (const float*)d_in[25];
  const float* mix_b = (const float*)d_in[26];
  const int* BOS = (const int*)d_in[27];

  float* out = (float*)d_out;
  float* dec32 = out;                       // [B*T, H]
  float* gen = out + Bz * Tz * Hz;          // [B*T, V]
  float* ext = gen + (long)Bz * Tz * Vz;    // [B*T, V]
  float* gt_out = ext + (long)Bz * Tz * Vz; // [B*T]

  size_t off = 0;
  auto alloc = [&](size_t bytes) -> void* {
    off = (off + 255) & ~(size_t)255;
    void* p = (char*)d_ws + off;
    off += bytes;
    return p;
  };
  const long nW = H3 * Hz;  // 786432
  const long nQ = Hz * Hz;  // 262144
  _Float16* wih0_16 = (_Float16*)alloc(nW * 2);
  _Float16* whh0_16 = (_Float16*)alloc(nW * 2);
  _Float16* wih1_16 = (_Float16*)alloc(nW * 2);
  _Float16* whh1_16 = (_Float16*)alloc(nW * 2);
  _Float16* wq16 = (_Float16*)alloc(2 * nQ * 2);
  _Float16* wk16 = (_Float16*)alloc(2 * nQ * 2);
  _Float16* w1_16 = (_Float16*)alloc((long)Hz * 2048 * 2);
  _Float16* w2_16 = (_Float16*)alloc((long)Vz * Hz * 2);
  _Float16* m0_16 = (_Float16*)alloc((long)Bz * Lz * Hz * 2);
  _Float16* m1_16 = (_Float16*)alloc((long)Bz * Lz * Hz * 2);
  _Float16* m0T16 = (_Float16*)alloc((long)Bz * Hz * Lz * 2);
  _Float16* m1T16 = (_Float16*)alloc((long)Bz * Hz * Lz * 2);
  float* x32 = (float*)alloc((long)Bz * Tz * Ez * 4);
  _Float16* x16 = (_Float16*)alloc((long)Bz * Tz * Ez * 2);
  float* xg = (float*)alloc((long)Bz * Tz * H3 * 4);  // reused layer0/layer1
  _Float16* x1_16 = (_Float16*)alloc((long)Bz * Tz * Hz * 2);
  _Float16* dec16 = (_Float16*)alloc((long)Bz * Tz * Hz * 2);
  float* qh = (float*)alloc(2L * Bz * Tz * Hz * 4);
  float* kh = (float*)alloc(2L * Bz * Lz * Hz * 4);
  float* sbuf = (float*)alloc(2L * Bz * Tz * Lz * 4);
  _Float16* abuf16 = (_Float16*)alloc(2L * Bz * Tz * Lz * 2);
  float* pbuf = (float*)alloc(2L * Bz * Tz * Lz * 4);
  float* ctx = (float*)alloc(2L * Bz * Tz * Hz * 4);
  _Float16* feat16 = (_Float16*)alloc((long)Bz * Tz * 2048 * 2);
  float* g1_32 = (float*)alloc((long)Bz * Tz * Hz * 4);
  _Float16* g1_16 = (_Float16*)alloc((long)Bz * Tz * Hz * 2);
  float* pm = (float*)alloc((long)Bz * Tz * 3 * 4);
  _Float16* wcat16 = (_Float16*)alloc((long)Bz * Tz * 2 * Lz * 2);
  if (off > ws_size) return;

  auto cvt = [&](const float* s, _Float16* d, long n) {
    int g = (int)((n + 255) / 256);
    if (g > 4096) g = 4096;
    cvt_kernel<<<g, 256, 0, stream>>>(s, d, n);
  };
  cvt(Wih0, wih0_16, nW);
  cvt(Whh0, whh0_16, nW);
  cvt(Wih1, wih1_16, nW);
  cvt(Whh1, whh1_16, nW);
  cvt(attn_Wq, wq16, 2 * nQ);
  cvt(attn_Wk, wk16, 2 * nQ);
  cvt(gen_W1, w1_16, (long)Hz * 2048);
  cvt(gen_W2, w2_16, (long)Vz * Hz);
  cvt(mem0, m0_16, (long)Bz * Lz * Hz);
  cvt(mem1, m1_16, (long)Bz * Lz * Hz);
  transpose_mem_kernel<<<Bz * Lz, 256, 0, stream>>>(mem0, m0T16);
  transpose_mem_kernel<<<Bz * Lz, 256, 0, stream>>>(mem1, m1T16);

  // Embedding
  embed_kernel<<<Bz * Tz, 256, 0, stream>>>(gt, BOS, emb, x32, x16);

  auto gemm = [&](const _Float16* A, const _Float16* Bm, float* C,
                  _Float16* C16, const float* bias, int M, int N, int K,
                  int lda, int ldb, int ldc, long sA, long sB, long sC,
                  int batches) {
    dim3 grid((N / 16 + 7) / 8, M / 64, batches);
    gemm_f16_kernel<<<grid, 256, 0, stream>>>(A, Bm, C, C16, bias, M, N, K,
                                              lda, ldb, ldc, sA, sB, sC);
  };

  // GRU layer 0
  gemm(x16, wih0_16, xg, nullptr, bih0, Bz * Tz, H3, Ez, Ez, Ez, H3, 0, 0, 0, 1);
  gru_scan_kernel<<<1, 1024, 0, stream>>>(whh0_16, bhh0, xg, init_state,
                                          2 * Hz, x1_16, nullptr);
  // GRU layer 1
  gemm(x1_16, wih1_16, xg, nullptr, bih1, Bz * Tz, H3, Hz, Hz, Hz, H3, 0, 0, 0, 1);
  gru_scan_kernel<<<1, 1024, 0, stream>>>(whh1_16, bhh1, xg, init_state + Hz,
                                          2 * Hz, dec16, dec32);

  // Attention (two memories)
  for (int m = 0; m < 2; ++m) {
    const _Float16* memh = m ? m1_16 : m0_16;
    const _Float16* memT = m ? m1T16 : m0T16;
    gemm(dec16, wq16 + (long)m * nQ, qh + (long)m * Bz * Tz * Hz, nullptr,
         attn_bq + m * Hz, Bz * Tz, Hz, Hz, Hz, Hz, Hz, 0, 0, 0, 1);
    gemm(memh, wk16 + (long)m * nQ, kh + (long)m * Bz * Lz * Hz, nullptr,
         nullptr, Bz * Lz, Hz, Hz, Hz, Hz, Hz, 0, 0, 0, 1);
    attn_score_kernel<<<Bz * Tz * Lz / 8, 256, 0, stream>>>(
        qh + (long)m * Bz * Tz * Hz, kh + (long)m * Bz * Lz * Hz,
        attn_v + m * Hz, sbuf + (long)m * Bz * Tz * Lz);
    attn_softmax_kernel<<<Bz * Tz, Lz, 0, stream>>>(
        sbuf + (long)m * Bz * Tz * Lz, m ? mask1 : mask0, m ? w1 : w0,
        pbuf + (long)m * Bz * Tz * Lz, abuf16 + (long)m * Bz * Tz * Lz);
    // ctx = a @ mem : batched; B = memT [H, L] row-major ([N,K] fast path)
    gemm(abuf16 + (long)m * Bz * Tz * Lz, memT, ctx + (long)m * Bz * Tz * Hz,
         nullptr, nullptr, Tz, Hz, Lz, Lz, Lz, Hz, (long)Tz * Lz,
         (long)Hz * Lz, (long)Tz * Hz, Bz);
  }

  // feat / mix / generator
  feat_kernel<<<Bz * Tz, 256, 0, stream>>>(ctx, ctx + (long)Bz * Tz * Hz, x16,
                                           dec16, feat16);
  mix_kernel<<<Bz * Tz / 8, 256, 0, stream>>>(feat16, mix_W, mix_b, pm);
  gemm(feat16, w1_16, g1_32, g1_16, gen_b1, Bz * Tz, Hz, 2048, 2048, 2048, Hz,
       0, 0, 0, 1);
  gemm(g1_16, w2_16, gen, nullptr, nullptr, Bz * Tz, Vz, Hz, Hz, Hz, Vz, 0, 0,
       0, 1);
  row_softmax_kernel<<<Bz * Tz, 256, 0, stream>>>(gen);

  // copy distribution + final mix
  wcat_kernel<<<Bz * Tz, 2 * Lz, 0, stream>>>(pbuf, pbuf + (long)Bz * Tz * Lz,
                                              pm, wcat16);
  {
    dim3 grid(Vz / 16 / 8, 1, Bz);
    dist2_ext_kernel<<<grid, 256, 0, stream>>>(wcat16, smap, pm, gen, ext);
  }
  gt_copy_kernel<<<2, 256, 0, stream>>>(gt, gt_out);
}
